// ConnectionBlock_39745627357355
// MI455X (gfx1250) — matrix operand order repaired and checked by reading, not compiled, not run on hardware
//
#include <hip/hip_runtime.h>
#include <hip/hip_bf16.h>

typedef __attribute__((ext_vector_type(16))) _Float16 v16h;
typedef __attribute__((ext_vector_type(8)))  _Float16 v8h;
typedef __attribute__((ext_vector_type(8)))  float    v8f;

#define T_DIM 128
#define B_DIM 256
#define C_DIM 512
#define M_DIM (T_DIM * B_DIM)      // 32768 rows for the GEMM
#define BC    (B_DIM * C_DIM)      // 131072 LIF cells
#define STAT_BLOCKS 256
#define STAT_ROWS   (M_DIM / STAT_BLOCKS)  // 128

// ---------------- f32 -> f16 weight conversion ----------------
__global__ void cvt_f32_f16_kernel(const float* __restrict__ src,
                                   _Float16* __restrict__ dst, int n) {
  int i = blockIdx.x * blockDim.x + threadIdx.x;
  if (i < n) dst[i] = (_Float16)src[i];
}

// ---------------- zero small stats buffer ----------------
__global__ void zero_kernel(float* __restrict__ p, int n) {
  int i = blockIdx.x * blockDim.x + threadIdx.x;
  if (i < n) p[i] = 0.0f;
}

// ---------------- LIF scan: v = v*(1-1/tau) + x_t ; spike ; hard reset ----
__global__ void lif_kernel(const float* __restrict__ x,
                           const float* __restrict__ w,   // scalar logit
                           _Float16* __restrict__ spikes) {
  int idx = blockIdx.x * blockDim.x + threadIdx.x;
  if (idx >= BC) return;
  float inv_tau = 1.0f / (1.0f + expf(-w[0]));
  float keep = 1.0f - inv_tau;
  float v = 0.0f;
  for (int t = 0; t < T_DIM; ++t) {
    float xt = x[(size_t)t * BC + idx];
    v = v * keep + xt;
    float s = (v >= 1.0f) ? 1.0f : 0.0f;
    spikes[(size_t)t * BC + idx] = (_Float16)s;
    v = (1.0f - s) * v;  // hard reset to 0
  }
}

// ---------------- fragment loader for one 32-wide K step -------------------
// A (16x32, contiguous k order used here, same as B): lanes 0-15 hold K k0..k0+15,
// lanes 16-31 hold K k0+16..k0+31.
// B (32x16): lane half 0 -> K k0..k0+15, half 1 -> k0+16..k0+31, col = l.
__device__ __forceinline__ void load_frags(const _Float16* __restrict__ arow0,
                                           const _Float16* __restrict__ arow1,
                                           const _Float16* __restrict__ brow,
                                           int k0, int half,
                                           v16h& a0, v16h& a1, v16h* b) {
  v8h lo0 = *(const v8h*)(arow0 + k0 + 16 * half);
  v8h hi0 = *(const v8h*)(arow0 + k0 + 8 + 16 * half);
  v8h lo1 = *(const v8h*)(arow1 + k0 + 16 * half);
  v8h hi1 = *(const v8h*)(arow1 + k0 + 8 + 16 * half);
#pragma unroll
  for (int i = 0; i < 8; ++i) {
    a0[i] = lo0[i]; a0[i + 8] = hi0[i];
    a1[i] = lo1[i]; a1[i + 8] = hi1[i];
  }
#pragma unroll
  for (int jn = 0; jn < 4; ++jn) {
    b[jn] = *(const v16h*)(brow + (size_t)jn * 16 * C_DIM + k0 + 16 * half);
  }
}

// ---------------- WMMA GEMM: Y[M,N] = A[M,K] * B[N,K]^T + bias[N] ----------
// Each wave computes a 32(M) x 64(N) strip: 2x4 accumulator tiles.
// K loop is software-pipelined (double-buffered fragments) so next-step loads
// are in flight while the current 8 WMMAs execute.
__global__ __launch_bounds__(256)
void gemm_wmma_kernel(const _Float16* __restrict__ A,
                      const _Float16* __restrict__ Bm,   // [N,K] f16
                      const float* __restrict__ bias,
                      float* __restrict__ Y) {
  const int K = C_DIM, N = C_DIM;
  const int NGROUPS = N / 64;                                  // 8
  int wave = (blockIdx.x * blockDim.x + threadIdx.x) >> 5;
  int lane = threadIdx.x & 31;
  int half = lane >> 4;
  int l    = lane & 15;

  int m0 = (wave / NGROUPS) * 32;
  int n0 = (wave % NGROUPS) * 64;

  v8f zero = {0.f, 0.f, 0.f, 0.f, 0.f, 0.f, 0.f, 0.f};
  v8f acc[8] = {zero, zero, zero, zero, zero, zero, zero, zero};

  const _Float16* arow0 = A + (size_t)(m0 + l) * K;
  const _Float16* arow1 = A + (size_t)(m0 + 16 + l) * K;
  const _Float16* brow  = Bm + (size_t)(n0 + l) * K;

  v16h a0c, a1c, bc[4];
  load_frags(arow0, arow1, brow, 0, half, a0c, a1c, bc);

#pragma unroll 4
  for (int k0 = 0; k0 < K; k0 += 32) {
    // Prefetch next step's fragments (wraps to 0 on the final iteration;
    // those loads are discarded — keeps the loop branch-free).
    v16h a0n, a1n, bn_[4];
    int kn = (k0 + 32) & (K - 1);
    load_frags(arow0, arow1, brow, kn, half, a0n, a1n, bn_);

#pragma unroll
    for (int jn = 0; jn < 4; ++jn) {
      acc[jn] = __builtin_amdgcn_wmma_f32_16x16x32_f16(
          false, a0c, false, bc[jn], (short)0, acc[jn], false, false);
      acc[4 + jn] = __builtin_amdgcn_wmma_f32_16x16x32_f16(
          false, a1c, false, bc[jn], (short)0, acc[4 + jn], false, false);
    }

    a0c = a0n; a1c = a1n;
#pragma unroll
    for (int jn = 0; jn < 4; ++jn) bc[jn] = bn_[jn];
  }

  // C/D layout: VGPR r holds row (16*jm + r + 8*half), column n0 + 16*jn + l.
#pragma unroll
  for (int jn = 0; jn < 4; ++jn) {
    int n = n0 + 16 * jn + l;
    float bn = bias[n];
#pragma unroll
    for (int jm = 0; jm < 2; ++jm) {
#pragma unroll
      for (int r = 0; r < 8; ++r) {
        Y[(size_t)(m0 + 16 * jm + r + 8 * half) * N + n] = acc[4 * jm + jn][r] + bn;
      }
    }
  }
}

// ---------------- BN stats: per-channel sum / sumsq over M rows ------------
__global__ void bn_stats_kernel(const float* __restrict__ Y,
                                float* __restrict__ sum,
                                float* __restrict__ sumsq) {
  int c = threadIdx.x;                 // 0..511
  int row0 = blockIdx.x * STAT_ROWS;
  float s = 0.0f, s2 = 0.0f;
  for (int r = 0; r < STAT_ROWS; ++r) {
    float v = Y[(size_t)(row0 + r) * C_DIM + c];
    s += v;
    s2 += v * v;
  }
  atomicAdd(&sum[c], s);
  atomicAdd(&sumsq[c], s2);
}

// ---------------- BN apply (+ optional skip): out = norm(Y)*g + b [+ skip] --
__global__ void bn_apply_kernel(const float* __restrict__ Y,
                                const float* __restrict__ sum,
                                const float* __restrict__ sumsq,
                                const float* __restrict__ gamma,
                                const float* __restrict__ beta,
                                const float* __restrict__ skip,  // may be null
                                float* __restrict__ out) {
  size_t i = (size_t)blockIdx.x * blockDim.x + threadIdx.x;
  const size_t total = (size_t)M_DIM * C_DIM;
  if (i >= total) return;
  int c = (int)(i & (C_DIM - 1));
  const float invM = 1.0f / (float)M_DIM;
  float mean = sum[c] * invM;
  float var = sumsq[c] * invM - mean * mean;
  float inv = rsqrtf(var + 1e-5f);
  float v = (Y[i] - mean) * inv * gamma[c] + beta[c];
  if (skip) v += skip[i];
  out[i] = v;
}

extern "C" void kernel_launch(void* const* d_in, const int* in_sizes, int n_in,
                              void* d_out, int out_size, void* d_ws, size_t ws_size,
                              hipStream_t stream) {
  (void)in_sizes; (void)n_in; (void)out_size; (void)ws_size;
  const float* x      = (const float*)d_in[0];
  const float* w0     = (const float*)d_in[1];
  const float* W0     = (const float*)d_in[2];
  const float* b0     = (const float*)d_in[3];
  const float* gamma0 = (const float*)d_in[4];
  const float* beta0  = (const float*)d_in[5];
  const float* w1     = (const float*)d_in[6];
  const float* W1     = (const float*)d_in[7];
  const float* b1     = (const float*)d_in[8];
  const float* gamma1 = (const float*)d_in[9];
  const float* beta1  = (const float*)d_in[10];
  float* out = (float*)d_out;

  // Workspace layout
  char* ws = (char*)d_ws;
  _Float16* W16_0 = (_Float16*)ws;                       ws += (size_t)C_DIM * C_DIM * 2;
  _Float16* W16_1 = (_Float16*)ws;                       ws += (size_t)C_DIM * C_DIM * 2;
  _Float16* spk   = (_Float16*)ws;                       ws += (size_t)M_DIM * C_DIM * 2;
  float*    ybuf  = (float*)ws;                          ws += (size_t)M_DIM * C_DIM * 4;
  float*    s_sum = (float*)ws;                          ws += C_DIM * 4;
  float*    s_sq  = (float*)ws;                          ws += C_DIM * 4;

  const int WN = C_DIM * C_DIM;            // 262144 weights
  const size_t total = (size_t)M_DIM * C_DIM;
  const int gemm_waves = (M_DIM / 32) * (C_DIM / 64);   // 8192 waves
  const int gemm_blocks = gemm_waves / 8;               // 256 thr = 8 waves
  const int ew_blocks = (int)((total + 255) / 256);

  // Convert weights once per launch
  cvt_f32_f16_kernel<<<(WN + 255) / 256, 256, 0, stream>>>(W0, W16_0, WN);
  cvt_f32_f16_kernel<<<(WN + 255) / 256, 256, 0, stream>>>(W1, W16_1, WN);

  // ---- Block 0 ----
  zero_kernel<<<4, 256, 0, stream>>>(s_sum, C_DIM);
  zero_kernel<<<4, 256, 0, stream>>>(s_sq, C_DIM);
  lif_kernel<<<BC / 256, 256, 0, stream>>>(x, w0, spk);
  gemm_wmma_kernel<<<gemm_blocks, 256, 0, stream>>>(spk, W16_0, b0, ybuf);
  bn_stats_kernel<<<STAT_BLOCKS, C_DIM, 0, stream>>>(ybuf, s_sum, s_sq);
  bn_apply_kernel<<<ew_blocks, 256, 0, stream>>>(ybuf, s_sum, s_sq, gamma0, beta0,
                                                 nullptr, ybuf);

  // ---- Block 1 ----
  zero_kernel<<<4, 256, 0, stream>>>(s_sum, C_DIM);
  zero_kernel<<<4, 256, 0, stream>>>(s_sq, C_DIM);
  lif_kernel<<<BC / 256, 256, 0, stream>>>(ybuf, w1, spk);
  gemm_wmma_kernel<<<gemm_blocks, 256, 0, stream>>>(spk, W16_1, b1, out);
  bn_stats_kernel<<<STAT_BLOCKS, C_DIM, 0, stream>>>(out, s_sum, s_sq);
  bn_apply_kernel<<<ew_blocks, 256, 0, stream>>>(out, s_sum, s_sq, gamma1, beta1,
                                                 x, out);
}
